// Model_73658689126907
// MI455X (gfx1250) — matrix operand order repaired and checked by reading, not compiled, not run on hardware
//
#include <hip/hip_runtime.h>
#include <hip/hip_bf16.h>

typedef __attribute__((ext_vector_type(16))) _Float16 v16h;
typedef __attribute__((ext_vector_type(8)))  float    v8f;
typedef __attribute__((ext_vector_type(8)))  unsigned int v8u;

static constexpr int N_NODES = 100000;
static constexpr int N_EDGES = 1600000;
static constexpr int N_PREM  = 20000;
static constexpr int LAYERS  = 24;
static constexpr int LDO     = 400;          // O row stride (2K*(LAYERS+1))
static constexpr int NTILES  = N_NODES / 16; // 6250 exactly

// ---------------- utility kernels ----------------

__global__ void zeroF(float* p, int n) {
    int g = blockIdx.x * 256 + threadIdx.x;
    if (g < n) p[g] = 0.f;
}

__global__ void degK(const int* __restrict__ e0, const int* __restrict__ e1,
                     int* degIn, int* degOut) {
    int g = blockIdx.x * 256 + threadIdx.x;
    if (g < N_EDGES) {
        atomicAdd(&degIn[e1[g]], 1);   // in-direction: aggregation index = e1
        atomicAdd(&degOut[e0[g]], 1);  // out-direction: aggregation index = e0
    }
}

__global__ void dinvK(const int* degIn, const int* degOut,
                      float* dinvIn, float* dinvOut, float* selfIn, float* selfOut) {
    int n = blockIdx.x * 256 + threadIdx.x;
    if (n < N_NODES) {
        float di = (float)(degIn[n] + 1);   // +1 self loop
        float dq = (float)(degOut[n] + 1);
        dinvIn[n]  = rsqrtf(di);
        dinvOut[n] = rsqrtf(dq);
        selfIn[n]  = 1.f / di;   // dinv^2 for the self loop edge
        selfOut[n] = 1.f / dq;
    }
}

__global__ void normK(const int* __restrict__ e0, const int* __restrict__ e1,
                      const float* dinvIn, const float* dinvOut,
                      float* normIn, float* normOut) {
    int e = blockIdx.x * 256 + threadIdx.x;
    if (e < N_EDGES) {
        normIn[e]  = dinvIn[e0[e]]  * dinvIn[e1[e]];
        normOut[e] = dinvOut[e0[e]] * dinvOut[e1[e]];
    }
}

__global__ void embedK(const int* __restrict__ tok, const float* __restrict__ emb,
                       float* __restrict__ O) {
    int g = blockIdx.x * 256 + threadIdx.x;   // over N*16
    int n = g >> 4, j = g & 15;
    if (n < N_NODES) O[(size_t)n * LDO + j] = emb[tok[n] * 16 + j];
}

// ---------------- BN statistics ----------------
__global__ void statsRelu(const float* __restrict__ O, const int* __restrict__ rowIdx,
                          int nrows, float* sum, float* sumsq) {
    __shared__ float sd[256], sd2[256];
    int t = threadIdx.x;
    int fl = t & 15;
    int f = blockIdx.y * 16 + fl;
    float s = 0.f, s2 = 0.f;
    for (int r = blockIdx.x * 16 + (t >> 4); r < nrows; r += gridDim.x * 16) {
        int ri = rowIdx ? rowIdx[r] : r;
        float x = O[(size_t)ri * LDO + f];
        x = fmaxf(x, 0.f);
        s += x; s2 += x * x;
    }
    sd[t] = s; sd2[t] = s2;
    __syncthreads();
    for (int off = 128; off >= 16; off >>= 1) {
        if (t < off) { sd[t] += sd[t + off]; sd2[t] += sd2[t + off]; }
        __syncthreads();
    }
    if (t < 16) {
        atomicAdd(&sum[blockIdx.y * 16 + t], sd[t]);
        atomicAdd(&sumsq[blockIdx.y * 16 + t], sd2[t]);
    }
}

__global__ void finalizeBN(const float* sum, const float* sumsq,
                           const float* __restrict__ g, const float* __restrict__ b,
                           int nf, float invCnt, float* scale, float* shift) {
    int f = threadIdx.x;
    if (f < nf) {
        float mu  = sum[f] * invCnt;
        float var = sumsq[f] * invCnt - mu * mu;
        float rs  = rsqrtf(var + 1e-5f);
        float sc  = g[f] * rs;
        scale[f] = sc;
        shift[f] = b[f] - mu * sc;
    }
}

// ---------------- weight repack to WMMA f16 B-layout ----------------
// B (16-bit, 32x16): lane n = l%16, kbase = (l>=16?8:0), VGPR v holds K = kbase+2v+(v>=4?8:0), +1
__global__ void packW(const float* __restrict__ W, int cin, unsigned* __restrict__ out) {
    int chunk = blockIdx.x;
    int t = threadIdx.x;
    int v = t >> 5, lane = t & 31;
    int n = lane & 15;
    int k0 = chunk * 32 + ((lane >= 16) ? 8 : 0) + 2 * v + ((v >= 4) ? 8 : 0);
    int k1 = k0 + 1;
    float w0 = (n < 8 && k0 < cin) ? W[k0 * 8 + n] : 0.f;
    float w1 = (n < 8 && k1 < cin) ? W[k1 * 8 + n] : 0.f;
    _Float16 h0 = (_Float16)w0, h1 = (_Float16)w1;
    unsigned u = (unsigned)__builtin_bit_cast(unsigned short, h0) |
                 ((unsigned)__builtin_bit_cast(unsigned short, h1) << 16);
    out[(chunk * 8 + v) * 32 + lane] = u;
}

// BN(relu(x))*scale+shift for 8 consecutive features, packed into a[o..o+7] as f16.
__device__ __forceinline__ void bnPack8(v16h& a, const int o,
                                        const float* __restrict__ x,
                                        const float* __restrict__ s,
                                        const float* __restrict__ sh) {
    float4 xa = *reinterpret_cast<const float4*>(x);
    float4 xb = *reinterpret_cast<const float4*>(x + 4);
    float4 sa = *reinterpret_cast<const float4*>(s);
    float4 sb = *reinterpret_cast<const float4*>(s + 4);
    float4 ha = *reinterpret_cast<const float4*>(sh);
    float4 hb = *reinterpret_cast<const float4*>(sh + 4);
    a[o + 0] = (_Float16)fmaf(fmaxf(xa.x, 0.f), sa.x, ha.x);
    a[o + 1] = (_Float16)fmaf(fmaxf(xa.y, 0.f), sa.y, ha.y);
    a[o + 2] = (_Float16)fmaf(fmaxf(xa.z, 0.f), sa.z, ha.z);
    a[o + 3] = (_Float16)fmaf(fmaxf(xa.w, 0.f), sa.w, ha.w);
    a[o + 4] = (_Float16)fmaf(fmaxf(xb.x, 0.f), sb.x, hb.x);
    a[o + 5] = (_Float16)fmaf(fmaxf(xb.y, 0.f), sb.y, hb.y);
    a[o + 6] = (_Float16)fmaf(fmaxf(xb.z, 0.f), sb.z, hb.z);
    a[o + 7] = (_Float16)fmaf(fmaxf(xb.w, 0.f), sb.w, hb.w);
}

// ---------------- FC via WMMA f16 -> f32, fused BN(relu) input + conv-BN stats ----------------
// A (16-bit, 16x32): lane m holds row m; its 16 halves are two contiguous K-segments
// [kc+koff .. +7] and [kc+16+koff .. +7] with koff = (lane>=16)*8.  cin % 16 == 0, so the
// only partial chunk is a 16-wide tail where the high half (v>=4) is entirely invalid —
// a compile-time-uniform case, no lane-divergent guards needed.
__global__ void __launch_bounds__(256)
fcWmma(const float* __restrict__ O, int cin, const unsigned* __restrict__ Bp,
       const float* __restrict__ biasFC, const float* __restrict__ scl,
       const float* __restrict__ shf, float* __restrict__ H,
       float* hsum, float* hsq) {
    __shared__ float hs[8], hs2[8];
    int t = threadIdx.x;
    if (t < 8) { hs[t] = 0.f; hs2[t] = 0.f; }
    __syncthreads();

    int wave = t >> 5, lane = t & 31;
    int tile = blockIdx.x * 8 + wave;
    int m = lane & 15;
    int koff = (lane >= 16) ? 8 : 0;

    if (tile < NTILES) {
        const float* orow = O + (size_t)(tile * 16 + m) * LDO;
        v8f acc = {0.f, 0.f, 0.f, 0.f, 0.f, 0.f, 0.f, 0.f};
        int full = cin >> 5;
        for (int c = 0; c < full; c++) {
            int kb = c * 32 + koff;
            v16h a;
            bnPack8(a, 0, orow + kb,      scl + kb,      shf + kb);
            bnPack8(a, 8, orow + kb + 16, scl + kb + 16, shf + kb + 16);
            v8u bu;
#pragma unroll
            for (int v = 0; v < 8; v++) bu[v] = Bp[(c * 8 + v) * 32 + lane];
            v16h b = __builtin_bit_cast(v16h, bu);
            acc = __builtin_amdgcn_wmma_f32_16x16x32_f16(false, a, false, b,
                                                         (short)0, acc, false, false);
        }
        if (cin & 16) {               // uniform tail: only low K-half valid
            int kb = full * 32 + koff;
            v16h a;
            bnPack8(a, 0, orow + kb, scl + kb, shf + kb);
#pragma unroll
            for (int i = 8; i < 16; i++) a[i] = (_Float16)0.f;
            v8u bu;
#pragma unroll
            for (int v = 0; v < 8; v++) bu[v] = Bp[(full * 8 + v) * 32 + lane];
            v16h b = __builtin_bit_cast(v16h, bu);  // high half already zero-packed
            acc = __builtin_amdgcn_wmma_f32_16x16x32_f16(false, a, false, b,
                                                         (short)0, acc, false, false);
        }
        // D (f32 16x16): VGPR v, lane -> M = v + (lane>=16?8:0), N = lane%16 ; only N<8 real
        int n = m;
        if (n < 8) {
            float bv = biasFC[n];
#pragma unroll
            for (int v = 0; v < 8; v++) {
                int rm = tile * 16 + v + koff;   // koff==8 exactly when lane>=16
                float val = acc[v] + bv;
                H[(size_t)rm * 8 + n] = val;
                float rv = fmaxf(val, 0.f);
                atomicAdd(&hs[n], rv);
                atomicAdd(&hs2[n], rv * rv);
            }
        }
    }
    __syncthreads();
    if (t < 8) { atomicAdd(&hsum[t], hs[t]); atomicAdd(&hsq[t], hs2[t]); }
}

// ---------------- conv prep: BN(relu(H)) @ {Win,Wout}, init O cols w/ self-loop + bias ----------------
__global__ void convPrep(const float* __restrict__ H, const float* scl, const float* shf,
                         const float* __restrict__ Win, const float* __restrict__ bin,
                         const float* __restrict__ Wout, const float* __restrict__ bout,
                         const float* __restrict__ selfIn, const float* __restrict__ selfOut,
                         float* __restrict__ Gin, float* __restrict__ Gout,
                         float* __restrict__ O, int base) {
    __shared__ float Ws[128];
    int t = threadIdx.x;
    if (t < 64) Ws[t] = Win[t];
    else if (t < 128) Ws[t] = Wout[t - 64];
    __syncthreads();

    int n = blockIdx.x * 256 + t;
    if (n >= N_NODES) return;

    float h2[8];
#pragma unroll
    for (int j = 0; j < 8; j++)
        h2[j] = fmaxf(H[(size_t)n * 8 + j], 0.f) * scl[j] + shf[j];

    float sIn = selfIn[n], sOut = selfOut[n];
    float* orow = O + (size_t)n * LDO + base;
#pragma unroll
    for (int o = 0; o < 8; o++) {
        float si = 0.f, so = 0.f;
#pragma unroll
        for (int j = 0; j < 8; j++) {
            si = fmaf(h2[j], Ws[j * 8 + o], si);
            so = fmaf(h2[j], Ws[64 + j * 8 + o], so);
        }
        Gin[(size_t)n * 8 + o]  = si;
        Gout[(size_t)n * 8 + o] = so;
        orow[o]     = si * sIn  + bin[o];
        orow[8 + o] = so * sOut + bout[o];
    }
}

// ---------------- edge scatter: both directions, 8 lanes per edge ----------------
__global__ void scatterK(const int* __restrict__ e0, const int* __restrict__ e1,
                         const float* __restrict__ nIn, const float* __restrict__ nOut,
                         const float* __restrict__ Gin, const float* __restrict__ Gout,
                         float* __restrict__ O, int base) {
    int gid = blockIdx.x * 256 + threadIdx.x;   // over 2*E*8 = 25.6M
    int j = gid & 7;
    int tt = gid >> 3;
    if (tt < N_EDGES) {
        int e = tt;
        int s = e0[e], d = e1[e];
        atomicAdd(&O[(size_t)d * LDO + base + j], Gin[(size_t)s * 8 + j] * nIn[e]);
    } else if (tt < 2 * N_EDGES) {
        int e = tt - N_EDGES;
        int s = e1[e], d = e0[e];
        atomicAdd(&O[(size_t)d * LDO + base + 8 + j], Gout[(size_t)s * 8 + j] * nOut[e]);
    }
}

// ---------------- readout ----------------
__global__ void outK(const float* __restrict__ O, const int* __restrict__ pidx,
                     const float* scl, const float* shf,
                     const float* __restrict__ W, const float* __restrict__ bias,
                     float* __restrict__ out) {
    int p = blockIdx.x * 256 + threadIdx.x;
    if (p >= N_PREM) return;
    const float* row = O + (size_t)pidx[p] * LDO;
    float acc = bias[0];
    for (int f = 0; f < LDO; f++) {
        float x = fmaxf(row[f], 0.f) * scl[f] + shf[f];
        acc = fmaf(x, W[f], acc);
    }
    out[p] = acc;
}

// ---------------- host orchestration ----------------

extern "C" void kernel_launch(void* const* d_in, const int* in_sizes, int n_in,
                              void* d_out, int out_size, void* d_ws, size_t ws_size,
                              hipStream_t stream) {
    (void)in_sizes; (void)n_in; (void)out_size; (void)ws_size;

    // input pointers (setup_inputs() dict insertion order, recursively flattened)
    const float* embedP = (const float*)d_in[0];
    auto fcG   = [&](int l) { return (const float*)d_in[1 + 4 * l]; };
    auto fcB   = [&](int l) { return (const float*)d_in[2 + 4 * l]; };
    auto fcW   = [&](int l) { return (const float*)d_in[3 + 4 * l]; };
    auto fcBi  = [&](int l) { return (const float*)d_in[4 + 4 * l]; };
    auto cvG   = [&](int l) { return (const float*)d_in[97 + 6 * l + 0]; };
    auto cvB   = [&](int l) { return (const float*)d_in[97 + 6 * l + 1]; };
    auto cvWin = [&](int l) { return (const float*)d_in[97 + 6 * l + 2]; };
    auto cvBin = [&](int l) { return (const float*)d_in[97 + 6 * l + 3]; };
    auto cvWo  = [&](int l) { return (const float*)d_in[97 + 6 * l + 4]; };
    auto cvBo  = [&](int l) { return (const float*)d_in[97 + 6 * l + 5]; };
    const float* outG  = (const float*)d_in[241];
    const float* outB  = (const float*)d_in[242];
    const float* outW  = (const float*)d_in[243];
    const float* outBi = (const float*)d_in[244];
    const int* tok  = (const int*)d_in[245];
    const int* e0   = (const int*)d_in[246];
    const int* e1   = e0 + N_EDGES;
    const int* pidx = (const int*)d_in[247];
    float* outp = (float*)d_out;

    // workspace layout (floats)
    float* ws = (float*)d_ws;
    float*  O       = ws;                                  // N*400
    float*  H       = O + (size_t)N_NODES * LDO;           // N*8
    float*  Gin     = H + (size_t)N_NODES * 8;             // N*8
    float*  Gout    = Gin + (size_t)N_NODES * 8;           // N*8
    float*  dinvIn  = Gout + (size_t)N_NODES * 8;          // N
    float*  dinvOut = dinvIn + N_NODES;                    // N
    float*  selfIn  = dinvOut + N_NODES;                   // N
    float*  selfOut = selfIn + N_NODES;                    // N
    float*  normIn  = selfOut + N_NODES;                   // E
    float*  normOut = normIn + N_EDGES;                    // E
    float*  sum     = normOut + N_EDGES;                   // 512
    float*  sumsq   = sum + 512;                           // 512
    float*  scale   = sumsq + 512;                         // 512
    float*  shift   = scale + 512;                         // 512
    int*    degIn   = (int*)(shift + 512);                 // N ints
    int*    degOut  = degIn + N_NODES;                     // N ints
    unsigned* Bpack = (unsigned*)(degOut + N_NODES);       // 24 * 12 * 256 uints

    // ---- setup ----
    zeroF<<<(2 * N_NODES + 255) / 256, 256, 0, stream>>>((float*)degIn, 2 * N_NODES);
    degK<<<N_EDGES / 256, 256, 0, stream>>>(e0, e1, degIn, degOut);
    dinvK<<<(N_NODES + 255) / 256, 256, 0, stream>>>(degIn, degOut, dinvIn, dinvOut, selfIn, selfOut);
    normK<<<N_EDGES / 256, 256, 0, stream>>>(e0, e1, dinvIn, dinvOut, normIn, normOut);
    embedK<<<N_NODES * 16 / 256, 256, 0, stream>>>(tok, embedP, O);

    // ---- layers ----
    for (int l = 0; l < LAYERS; l++) {
        int cin = 16 * (l + 1);
        int base = 16 * (l + 1);
        int chunks = (cin + 31) / 32;
        unsigned* Bp = Bpack + (size_t)l * 12 * 256;

        zeroF<<<4, 256, 0, stream>>>(sum, 1024);  // zeros sum[512] + sumsq[512]
        statsRelu<<<dim3(32, cin / 16), 256, 0, stream>>>(O, nullptr, N_NODES, sum, sumsq);
        finalizeBN<<<1, 512, 0, stream>>>(sum, sumsq, fcG(l), fcB(l), cin,
                                          1.f / N_NODES, scale, shift);
        packW<<<chunks, 256, 0, stream>>>(fcW(l), cin, Bp);
        fcWmma<<<(NTILES + 7) / 8, 256, 0, stream>>>(O, cin, Bp, fcBi(l), scale, shift,
                                                     H, sum + 448, sumsq + 448);
        finalizeBN<<<1, 512, 0, stream>>>(sum + 448, sumsq + 448, cvG(l), cvB(l), 8,
                                          1.f / N_NODES, scale + 448, shift + 448);
        convPrep<<<(N_NODES + 255) / 256, 256, 0, stream>>>(H, scale + 448, shift + 448,
                                                            cvWin(l), cvBin(l), cvWo(l), cvBo(l),
                                                            selfIn, selfOut, Gin, Gout, O, base);
        scatterK<<<2 * N_EDGES * 8 / 256, 256, 0, stream>>>(e0, e1, normIn, normOut,
                                                            Gin, Gout, O, base);
    }

    // ---- readout over premise nodes ----
    zeroF<<<4, 256, 0, stream>>>(sum, 1024);
    statsRelu<<<dim3(32, LDO / 16), 256, 0, stream>>>(O, pidx, N_PREM, sum, sumsq);
    finalizeBN<<<1, 512, 0, stream>>>(sum, sumsq, outG, outB, LDO, 1.f / N_PREM, scale, shift);
    outK<<<(N_PREM + 255) / 256, 256, 0, stream>>>(O, pidx, scale, shift, outW, outBi, outp);
}
